// MutualRecModel_9216999817732
// MI455X (gfx1250) — matrix operand order repaired and checked by reading, compiled, not run on hardware
//
#include <hip/hip_runtime.h>
#include <hip/hip_bf16.h>

// ---------------- problem constants (from reference) ----------------
#define PREDN   100000
#define TOTALN  120000
#define ITEMN   100000
#define NUIN    200000   // PRED + ITEM
#define NODESN  220000   // TOTAL + ITEM
#define EMBN    64

typedef __attribute__((ext_vector_type(16))) _Float16 v16h;
typedef __attribute__((ext_vector_type(8)))  _Float16 v8h;
typedef __attribute__((ext_vector_type(8)))  float    v8f;

// GEMM flags
#define GF_ACC   1   // accumulate into existing Y
#define GF_BIAS  2   // add bias[col]
#define GF_LEAKY 4   // leaky_relu(0.01)

static inline int divup_i(long a, long b) { return (int)((a + b - 1) / b); }

// =====================================================================
// Operand preparation for WMMA GEMM
// =====================================================================
// X (f32, possibly split concat) -> Xh (f16 row-major, n x K)
__global__ void cvt_x(const float* __restrict__ X1, const float* __restrict__ X2,
                      _Float16* __restrict__ Xh, long n, int kshift)
{
  long i = (long)blockIdx.x * blockDim.x + threadIdx.x;
  const int K = 1 << kshift;
  if (i >= (n << kshift)) return;
  int k = (int)(i & (K - 1));
  long r = i >> kshift;
  float v = (k < 64) ? X1[r * 64 + k] : X2[r * 64 + (k - 64)];
  Xh[i] = (_Float16)v;
}

// W (f32, K x 64 row-major) -> packed f16 B-fragments:
// Bp[(((kc*4 + t)*32 + lane)*16 + e], k = 8*(lane>>4) + e (e<8), 16 + 8*(lane>>4) + e-8 (e>=8).
__global__ void pack_w(const float* __restrict__ W, _Float16* __restrict__ Bp, int K)
{
  int i = (int)((long)blockIdx.x * blockDim.x + threadIdx.x);
  if (i >= K * 64) return;
  int e    = i & 15;
  int lane = (i >> 4) & 31;
  int t    = (i >> 9) & 3;
  int kc   = i >> 11;
  int v = e >> 1, h = e & 1;
  int kk  = kc * 32 + (lane >> 4) * 8 + (e & 7) + ((e >> 3) << 4); (void)v; (void)h;
  int col = t * 16 + (lane & 15);
  Bp[i] = (_Float16)W[(long)kk * 64 + col];
}

// =====================================================================
// WMMA GEMM:  Y[n x 64] (+)= Xh[n x K](f16) @ W[K x 64] (+ bias) (leaky)
// Packed B fragments staged in LDS once per block; one wave = 16x64 tile.
// =====================================================================
__global__ __launch_bounds__(256)
void gemm_wmma_f16p(const _Float16* __restrict__ Xh, int n, int K,
                    const _Float16* __restrict__ Bp,
                    const float* __restrict__ bias,
                    float* __restrict__ Y, int flags)
{
  __shared__ __align__(16) _Float16 sB[128 * 64];   // up to 16 KB of packed W

  // cooperative staging (all threads participate; barrier before any exit)
  const int nh = K * 64;
  for (int o = threadIdx.x * 8; o < nh; o += 256 * 8)
    *(v8h*)(sB + o) = *(const v8h*)(Bp + o);
  __syncthreads();

  const int lane = threadIdx.x & 31;
  const int wave = threadIdx.x >> 5;
  const int rowBase = (blockIdx.x * 8 + wave) * 16;
  if (rowBase >= n) return;                 // wave-uniform: EXEC all-1s at WMMA
  const int hi  = lane >> 4;
  const int l15 = lane & 15;

  v8f acc[4] = {};
  const _Float16* xrow = Xh + (long)(rowBase + l15) * K;

  for (int kc = 0; kc < (K >> 5); ++kc) {
    const int k0 = kc * 32;
    // A fragment: two contiguous 8-half runs per lane (ISA 16-bit A layout)
    v8h alo = *(const v8h*)(xrow + k0 + hi * 8);
    v8h ahi = *(const v8h*)(xrow + k0 + 16 + hi * 8);
    v16h a = __builtin_shufflevector(alo, ahi,
                                     0,1,2,3,4,5,6,7,8,9,10,11,12,13,14,15);
#pragma unroll
    for (int t = 0; t < 4; ++t) {
      v16h b = *(const v16h*)(sB + (((kc * 4 + t) * 32 + lane) << 4));
      acc[t] = __builtin_amdgcn_wmma_f32_16x16x32_f16(
          false, a, false, b, (short)0, acc[t], false, false);
    }
  }

  // store per C/D layout: VGPR r -> row rowBase + r + hi*8
#pragma unroll
  for (int t = 0; t < 4; ++t) {
#pragma unroll
    for (int r = 0; r < 8; ++r) {
      const int row = rowBase + r + hi * 8;
      const int col = t * 16 + l15;
      const long idx = (long)row * 64 + col;
      float v = acc[t][r];
      if (flags & GF_ACC)   v += Y[idx];
      if (flags & GF_BIAS)  v += bias[col];
      if (flags & GF_LEAKY) v = v > 0.f ? v : 0.01f * v;
      Y[idx] = v;
    }
  }
}

// =====================================================================
// Elementwise / fill / copy
// =====================================================================
__global__ void fill_f(float* __restrict__ p, float v, long n) {
  long i = (long)blockIdx.x * blockDim.x + threadIdx.x;
  if (i < n) p[i] = v;
}
__global__ void leaky_inplace(float* __restrict__ p, long n) {
  long i = (long)blockIdx.x * blockDim.x + threadIdx.x;
  if (i < n) { float v = p[i]; p[i] = v > 0.f ? v : 0.01f * v; }
}
__global__ void mul_ew(const float* __restrict__ a, const float* __restrict__ b,
                       float* __restrict__ o, long n) {
  long i = (long)blockIdx.x * blockDim.x + threadIdx.x;
  if (i < n) o[i] = a[i] * b[i];
}
// uie = concat(e[:PRED], e[TOTAL:TOTAL+ITEM])
__global__ void build_uie(const float* __restrict__ e, float* __restrict__ uie) {
  long i = (long)blockIdx.x * blockDim.x + threadIdx.x;
  if (i >= (long)NUIN * 64) return;
  int r = (int)(i >> 6), c = (int)(i & 63);
  int er = (r < PREDN) ? r : (TOTALN + (r - PREDN));
  uie[i] = e[(long)er * 64 + c];
}

// =====================================================================
// BatchNorm: column stats (64 cols) -> scale/shift -> apply
// stats layout: [0..63]=sum, [64..127]=sumsq, [128..191]=scale, [192..255]=shift
// =====================================================================
__global__ void bn_stats(const float* __restrict__ Y, float* __restrict__ stats, int n) {
  long g = (long)blockIdx.x * blockDim.x + threadIdx.x;
  int col = (int)(g & 63);
  long stride = ((long)gridDim.x * blockDim.x) >> 6;
  float s = 0.f, sq = 0.f;
  for (long r = g >> 6; r < n; r += stride) {
    float v = Y[r * 64 + col];
    s += v; sq += v * v;
  }
  atomicAdd(&stats[col], s);
  atomicAdd(&stats[64 + col], sq);
}
__global__ void bn_finalize(const float* __restrict__ g, const float* __restrict__ b,
                            float* __restrict__ stats, int n) {
  int c = threadIdx.x;
  if (c >= 64) return;
  float inv = 1.f / (float)n;
  float mean = stats[c] * inv;
  float var = fmaxf(stats[64 + c] * inv - mean * mean, 0.f);
  float sc = rsqrtf(var + 1e-5f) * g[c];
  stats[128 + c] = sc;
  stats[192 + c] = b[c] - mean * sc;
}
__global__ void bn_apply(const float* __restrict__ Y, const float* __restrict__ stats,
                         float* __restrict__ dst, long n64, int leaky) {
  long i = (long)blockIdx.x * blockDim.x + threadIdx.x;
  if (i >= n64) return;
  int col = (int)(i & 63);
  float v = Y[i] * stats[128 + col] + stats[192 + col];
  if (leaky) v = v > 0.f ? v : 0.01f * v;
  dst[i] = v;
}

// =====================================================================
// GATv2 edge kernels (wave32 per edge)
// =====================================================================
__device__ __forceinline__ void atomicMaxF(float* addr, float val) {
  if (val >= 0.f) atomicMax((int*)addr, __float_as_int(val));
  else            atomicMin((unsigned int*)addr, __float_as_uint(val));
}

__global__ void gat_edge_logits(const int* __restrict__ src, const int* __restrict__ dst,
                                const float* __restrict__ fs, const float* __restrict__ fd,
                                const float* __restrict__ attn,
                                float* __restrict__ ev, float* __restrict__ m, int nE)
{
  int wid = (int)(((long)blockIdx.x * blockDim.x + threadIdx.x) >> 5);
  int lane = threadIdx.x & 31;
  if (wid >= nE) return;
  int s = src[wid], d = dst[wid];
  // float2 per lane: coalesced b64 loads
  const float2* ps = (const float2*)(fs + (long)s * 64);
  const float2* pd = (const float2*)(fd + (long)d * 64);
  const float2* pa = (const float2*)attn;
  float2 a = ps[lane], b = pd[lane], w = pa[lane];
  float v0 = a.x + b.x, v1 = a.y + b.y;
  v0 = v0 > 0.f ? v0 : 0.2f * v0;
  v1 = v1 > 0.f ? v1 : 0.2f * v1;
  float acc = v0 * w.x + v1 * w.y;
#pragma unroll
  for (int off = 16; off > 0; off >>= 1) acc += __shfl_xor(acc, off, 32);
  if (lane == 0) { ev[wid] = acc; atomicMaxF(&m[d], acc); }
}

__global__ void gat_edge_exp(const int* __restrict__ dst, const float* __restrict__ ev,
                             const float* __restrict__ m, float* __restrict__ ex,
                             float* __restrict__ denom, int nE)
{
  int i = (int)((long)blockIdx.x * blockDim.x + threadIdx.x);
  if (i >= nE) return;
  int d = dst[i];
  float e = __expf(ev[i] - m[d]);
  ex[i] = e;
  atomicAdd(&denom[d], e);
}

__global__ void gat_edge_scatter(const int* __restrict__ src, const int* __restrict__ dst,
                                 const float* __restrict__ fs, const float* __restrict__ ex,
                                 const float* __restrict__ denom, float* __restrict__ out, int nE)
{
  int wid = (int)(((long)blockIdx.x * blockDim.x + threadIdx.x) >> 5);
  int lane = threadIdx.x & 31;
  if (wid >= nE) return;
  int s = src[wid], d = dst[wid];
  float alpha = ex[wid] / denom[d];
  const float2 a = ((const float2*)(fs + (long)s * 64))[lane];
  float* po = out + (long)d * 64 + 2 * lane;
  atomicAdd(po,     alpha * a.x);
  atomicAdd(po + 1, alpha * a.y);
}

// =====================================================================
// ChebConv kernels
// =====================================================================
__global__ void deg_count(const int* __restrict__ dst, float* __restrict__ deg, int nE) {
  int i = (int)((long)blockIdx.x * blockDim.x + threadIdx.x);
  if (i < nE) atomicAdd(&deg[dst[i]], 1.f);
}
__global__ void deg_rsqrt(float* __restrict__ deg, int n) {
  int i = (int)((long)blockIdx.x * blockDim.x + threadIdx.x);
  if (i < n) deg[i] = rsqrtf(fmaxf(deg[i], 1.f));   // dinv in-place
}
__global__ void cheb_scatter(const int* __restrict__ src, const int* __restrict__ dst,
                             const float* __restrict__ dinv, const float* __restrict__ X,
                             float* __restrict__ av, int nE)
{
  int wid = (int)(((long)blockIdx.x * blockDim.x + threadIdx.x) >> 5);
  int lane = threadIdx.x & 31;
  if (wid >= nE) return;
  int s = src[wid], d = dst[wid];
  float w = dinv[s] * dinv[d];
  const float2 a = ((const float2*)(X + (long)s * 64))[lane];
  float* pa = av + (long)d * 64 + 2 * lane;
  atomicAdd(pa,     w * a.x);
  atomicAdd(pa + 1, w * a.y);
}
__global__ void cheb_t1(const float* __restrict__ X, const float* __restrict__ av,
                        const float* __restrict__ lam, float* __restrict__ T1, long n) {
  long i = (long)blockIdx.x * blockDim.x + threadIdx.x;
  if (i >= n) return;
  float c = 2.f / lam[0];
  T1[i] = c * (X[i] - av[i]) - X[i];
}
__global__ void cheb_t2(const float* __restrict__ X, const float* __restrict__ T1,
                        const float* __restrict__ av2, const float* __restrict__ lam,
                        float* __restrict__ T2, long n) {
  long i = (long)blockIdx.x * blockDim.x + threadIdx.x;
  if (i >= n) return;
  float c = 2.f / lam[0];
  T2[i] = 2.f * (c * (T1[i] - av2[i]) - T1[i]) - X[i];
}

// =====================================================================
// soc = where(rowsum(i2u)!=0, i2u, uie[:PRED])   (wave per row)
// =====================================================================
__global__ void soc_select(const float* __restrict__ i2u, const float* __restrict__ uieP,
                           float* __restrict__ soc, int pred)
{
  int row = (int)(((long)blockIdx.x * blockDim.x + threadIdx.x) >> 5);
  int lane = threadIdx.x & 31;
  if (row >= pred) return;
  const float2 a = ((const float2*)(i2u + (long)row * 64))[lane];
  float s = a.x + a.y;
#pragma unroll
  for (int off = 16; off > 0; off >>= 1) s += __shfl_xor(s, off, 32);
  bool use = (s != 0.f);
  const float2 u = ((const float2*)(uieP + (long)row * 64))[lane];
  float2 r; r.x = use ? a.x : u.x; r.y = use ? a.y : u.y;
  ((float2*)(soc + (long)row * 64))[lane] = r;
}

// out = hm * softmax(h, axis=1)   (wave per row, 64 cols)
__global__ void softmax_mul(const float* __restrict__ h, const float* __restrict__ hm,
                            float* __restrict__ out, int pred)
{
  int row = (int)(((long)blockIdx.x * blockDim.x + threadIdx.x) >> 5);
  int lane = threadIdx.x & 31;
  if (row >= pred) return;
  const float2 v = ((const float2*)(h + (long)row * 64))[lane];
  float mx = fmaxf(v.x, v.y);
#pragma unroll
  for (int off = 16; off > 0; off >>= 1) mx = fmaxf(mx, __shfl_xor(mx, off, 32));
  float e0 = __expf(v.x - mx), e1 = __expf(v.y - mx);
  float s = e0 + e1;
#pragma unroll
  for (int off = 16; off > 0; off >>= 1) s += __shfl_xor(s, off, 32);
  const float2 m2 = ((const float2*)(hm + (long)row * 64))[lane];
  float2 r; r.x = m2.x * (e0 / s); r.y = m2.y * (e1 / s);
  ((float2*)(out + (long)row * 64))[lane] = r;
}

// =====================================================================
// Host-side helpers
// =====================================================================
static void run_gemm(hipStream_t st, const float* X1, const float* X2, int K, int n,
                     const float* W, const float* bias, float* Y, int flags,
                     _Float16* Xh, _Float16* Bp, int cvt /*1 = re-convert X*/)
{
  const int kshift = (K == 64) ? 6 : 7;
  if (cvt)
    cvt_x<<<divup_i((long)n * K, 256), 256, 0, st>>>(X1, X2, Xh, n, kshift);
  pack_w<<<divup_i(K * 64, 256), 256, 0, st>>>(W, Bp, K);
  gemm_wmma_f16p<<<divup_i(n, 128), 256, 0, st>>>(Xh, n, K, Bp, bias, Y, flags);
}

static void run_gat(hipStream_t st, const float* X, int n,
                    const int* src, const int* dst, int nE,
                    const float* Wl, const float* bl,
                    const float* Wr, const float* br, const float* attn,
                    float* fs, float* fd, float* out,
                    float* ev, float* ex, float* m, float* denom,
                    _Float16* Xh, _Float16* Bp)
{
  run_gemm(st, X, nullptr, 64, n, Wl, bl, fs, GF_BIAS, Xh, Bp, 1);
  run_gemm(st, X, nullptr, 64, n, Wr, br, fd, GF_BIAS, Xh, Bp, 0); // reuse Xh
  fill_f<<<divup_i(n, 256), 256, 0, st>>>(m, -3.0e38f, n);
  fill_f<<<divup_i(n, 256), 256, 0, st>>>(denom, 0.f, n);
  fill_f<<<divup_i((long)n * 64, 256), 256, 0, st>>>(out, 0.f, (long)n * 64);
  gat_edge_logits<<<divup_i((long)nE * 32, 256), 256, 0, st>>>(src, dst, fs, fd, attn, ev, m, nE);
  gat_edge_exp<<<divup_i(nE, 256), 256, 0, st>>>(dst, ev, m, ex, denom, nE);
  gat_edge_scatter<<<divup_i((long)nE * 32, 256), 256, 0, st>>>(src, dst, fs, ex, denom, out, nE);
  leaky_inplace<<<divup_i((long)n * 64, 256), 256, 0, st>>>(out, (long)n * 64);
}

static void run_mlp(hipStream_t st, const float* X1, const float* X2, int n,
                    const float* W, const float* b, const float* g, const float* beta,
                    float* ytmp, float* stats, float* dst, _Float16* Xh, _Float16* Bp)
{
  run_gemm(st, X1, X2, 128, n, W, b, ytmp, GF_BIAS, Xh, Bp, 1);
  fill_f<<<1, 256, 0, st>>>(stats, 0.f, 128);
  bn_stats<<<256, 256, 0, st>>>(ytmp, stats, n);
  bn_finalize<<<1, 64, 0, st>>>(g, beta, stats, n);
  bn_apply<<<divup_i((long)n * 64, 256), 256, 0, st>>>(ytmp, stats, dst, (long)n * 64, 1);
}

static void run_cheb(hipStream_t st, const float* X,
                     const int* src, const int* dst, int nE, const float* dinv,
                     const float* chebW, const float* chebB, const float* lam,
                     float* av, float* T1, float* T2, float* outp,
                     _Float16* Xh, _Float16* Bp)
{
  const long n64 = (long)TOTALN * 64;
  fill_f<<<divup_i(n64, 256), 256, 0, st>>>(av, 0.f, n64);
  cheb_scatter<<<divup_i((long)nE * 32, 256), 256, 0, st>>>(src, dst, dinv, X, av, nE);
  cheb_t1<<<divup_i(n64, 256), 256, 0, st>>>(X, av, lam, T1, n64);
  fill_f<<<divup_i(n64, 256), 256, 0, st>>>(av, 0.f, n64);
  cheb_scatter<<<divup_i((long)nE * 32, 256), 256, 0, st>>>(src, dst, dinv, T1, av, nE);
  cheb_t2<<<divup_i(n64, 256), 256, 0, st>>>(X, T1, av, lam, T2, n64);
  run_gemm(st, X,  nullptr, 64, TOTALN, chebW,        nullptr, outp, 0,                         Xh, Bp, 1);
  run_gemm(st, T1, nullptr, 64, TOTALN, chebW + 4096, nullptr, outp, GF_ACC,                    Xh, Bp, 1);
  run_gemm(st, T2, nullptr, 64, TOTALN, chebW + 8192, chebB,   outp, GF_ACC | GF_BIAS | GF_LEAKY, Xh, Bp, 1);
}

// =====================================================================
// kernel_launch
// =====================================================================
extern "C" void kernel_launch(void* const* d_in, const int* in_sizes, int n_in,
                              void* d_out, int out_size, void* d_ws, size_t ws_size,
                              hipStream_t stream)
{
  const float* emb     = (const float*)d_in[0];
  const float* bn0_g   = (const float*)d_in[1];
  const float* bn0_b   = (const float*)d_in[2];
  const float* gat_Wl  = (const float*)d_in[3];   // (5,64,64)
  const float* gat_bl  = (const float*)d_in[4];   // (5,64)
  const float* gat_Wr  = (const float*)d_in[5];
  const float* gat_br  = (const float*)d_in[6];
  const float* gat_at  = (const float*)d_in[7];   // (5,64)
  const float* cheb_W  = (const float*)d_in[8];   // (3,64,64)
  const float* cheb_b  = (const float*)d_in[9];   // (64,)
  const float* mlp_W   = (const float*)d_in[10];  // (5,128,64)
  const float* mlp_b   = (const float*)d_in[11];
  const float* mlp_g   = (const float*)d_in[12];
  const float* mlp_be  = (const float*)d_in[13];
  const float* lam     = (const float*)d_in[14];
  const int* u2i_src   = (const int*)d_in[15]; int nU2I = in_sizes[15];
  const int* u2i_dst   = (const int*)d_in[16];
  const int* rc_src    = (const int*)d_in[17]; int nRC  = in_sizes[17];
  const int* rc_dst    = (const int*)d_in[18];
  const int* i2u_src   = (const int*)d_in[19]; int nI2U = in_sizes[19];
  const int* i2u_dst   = (const int*)d_in[20];
  // d_in[21] = i2u_ids = arange(PRED) -> identity, unused
  const int* sn_src    = (const int*)d_in[22]; int nSN  = in_sizes[22];
  const int* sn_dst    = (const int*)d_in[23];
  const int* snn_src   = (const int*)d_in[24]; int nSNN = in_sizes[24];
  const int* snn_dst   = (const int*)d_in[25];
  float* outP = (float*)d_out;                    // h_new_P
  float* outS = outP + (long)PREDN * 64;          // h_new_S

  // ----- workspace layout (bump allocator, 256B aligned) -----
  char* wsp = (char*)d_ws;
  auto alloc = [&](long nfloats) -> float* {
    float* p = (float*)wsp;
    wsp += ((nfloats * sizeof(float) + 255) & ~255ul);
    return p;
  };
  long maxE = nU2I; if (nRC > maxE) maxE = nRC; if (nSNN > maxE) maxE = nSNN;
  if (nI2U > maxE) maxE = nI2U; if (nSN > maxE) maxE = nSN;

  float* e      = alloc((long)NODESN * 64);  // BN'd embeddings (pred_u / total_u)
  float* uie    = alloc((long)NUIN * 64);
  float* fsbuf  = alloc((long)NUIN * 64);
  float* fdbuf  = alloc((long)NUIN * 64);
  float* goutA  = alloc((long)NUIN * 64);
  float* goutB  = alloc((long)NUIN * 64);
  float* socbuf = alloc((long)PREDN * 64);
  float* hbuf   = alloc((long)TOTALN * 64);
  float* uitem  = alloc((long)PREDN * 64);
  float* h_uP   = alloc((long)PREDN * 64);
  float* h_uS   = alloc((long)PREDN * 64);
  float* ev     = alloc(maxE);
  float* ex     = alloc(maxE);
  float* mbuf   = alloc(NUIN);
  float* denom  = alloc(NUIN);
  float* dinv   = alloc(TOTALN);
  float* stats  = alloc(256);
  _Float16* Xh  = (_Float16*)alloc(((long)NUIN * 64) / 2);  // f16 activations (max n*K halves)
  _Float16* Bp  = (_Float16*)alloc(128 * 64 / 2);           // packed f16 W fragments

  // ----- initial BN over emb -> e ; build uie -----
  fill_f<<<1, 256, 0, stream>>>(stats, 0.f, 128);
  bn_stats<<<256, 256, 0, stream>>>(emb, stats, NODESN);
  bn_finalize<<<1, 64, 0, stream>>>(bn0_g, bn0_b, stats, NODESN);
  bn_apply<<<divup_i((long)NODESN * 64, 256), 256, 0, stream>>>(emb, stats, e, (long)NODESN * 64, 0);
  build_uie<<<divup_i((long)NUIN * 64, 256), 256, 0, stream>>>(e, uie);

#define WL(i) (gat_Wl + (long)(i) * 4096)
#define BL(i) (gat_bl + (long)(i) * 64)
#define WR(i) (gat_Wr + (long)(i) * 4096)
#define BR(i) (gat_br + (long)(i) * 64)
#define AT(i) (gat_at + (long)(i) * 64)
#define MW(i) (mlp_W  + (long)(i) * 8192)
#define MB(i) (mlp_b  + (long)(i) * 64)
#define MG(i) (mlp_g  + (long)(i) * 64)
#define MBe(i) (mlp_be + (long)(i) * 64)

  // GAT0: u2i on uie -> goutA (u2i_emb)
  run_gat(stream, uie, NUIN, u2i_src, u2i_dst, nU2I,
          WL(0), BL(0), WR(0), BR(0), AT(0), fsbuf, fdbuf, goutA, ev, ex, mbuf, denom, Xh, Bp);
  // GAT1: rc on u2i_emb -> goutB (iie = goutB[:PRED])
  run_gat(stream, goutA, NUIN, rc_src, rc_dst, nRC,
          WL(1), BL(1), WR(1), BR(1), AT(1), fsbuf, fdbuf, goutB, ev, ex, mbuf, denom, Xh, Bp);
  // GAT2: i2u on su = uie[:PRED] -> hbuf (i2u_emb)
  run_gat(stream, uie, PREDN, i2u_src, i2u_dst, nI2U,
          WL(2), BL(2), WR(2), BR(2), AT(2), fsbuf, fdbuf, hbuf, ev, ex, mbuf, denom, Xh, Bp);
  // soc = where(rowsum(i2u_emb)!=0, i2u_emb, uie[:PRED])
  soc_select<<<divup_i((long)PREDN * 32, 256), 256, 0, stream>>>(hbuf, uie, socbuf, PREDN);
  // GAT3: sn on soc -> goutA (sie)
  run_gat(stream, socbuf, PREDN, sn_src, sn_dst, nSN,
          WL(3), BL(3), WR(3), BR(3), AT(3), fsbuf, fdbuf, goutA, ev, ex, mbuf, denom, Xh, Bp);
  // MLP0: concat(iie, sie) -> uitem
  run_mlp(stream, goutB, goutA, PREDN, MW(0), MB(0), MG(0), MBe(0), fsbuf, stats, uitem, Xh, Bp);

  // ----- ChebConv x2 on total_u = e[:TOTAL] over snn graph -----
  fill_f<<<divup_i(TOTALN, 256), 256, 0, stream>>>(dinv, 0.f, TOTALN);
  deg_count<<<divup_i(nSNN, 256), 256, 0, stream>>>(snn_dst, dinv, nSNN);
  deg_rsqrt<<<divup_i(TOTALN, 256), 256, 0, stream>>>(dinv, TOTALN);
  run_cheb(stream, e, snn_src, snn_dst, nSNN, dinv, cheb_W, cheb_b, lam,
           fsbuf, fdbuf, goutB, goutA, Xh, Bp);              // h1 -> goutA
  run_cheb(stream, goutA, snn_src, snn_dst, nSNN, dinv, cheb_W, cheb_b, lam,
           fsbuf, fdbuf, goutB, hbuf, Xh, Bp);               // h2 -> hbuf
  // GAT4: snn on h2 -> goutA (usoc = goutA[:PRED])
  run_gat(stream, hbuf, TOTALN, snn_src, snn_dst, nSNN,
          WL(4), BL(4), WR(4), BR(4), AT(4), fsbuf, fdbuf, goutA, ev, ex, mbuf, denom, Xh, Bp);

  // ----- fusion head -----
  run_mlp(stream, uitem, e, PREDN, MW(1), MB(1), MG(1), MBe(1), fsbuf, stats, h_uP, Xh, Bp);
  run_mlp(stream, goutA, e, PREDN, MW(2), MB(2), MG(2), MBe(2), fsbuf, stats, h_uS, Xh, Bp);
  mul_ew<<<divup_i((long)PREDN * 64, 256), 256, 0, stream>>>(h_uP, h_uS, goutB, (long)PREDN * 64); // h_m
  softmax_mul<<<divup_i((long)PREDN * 32, 256), 256, 0, stream>>>(h_uP, goutB, socbuf, PREDN);
  run_mlp(stream, socbuf, h_uP, PREDN, MW(3), MB(3), MG(3), MBe(3), fsbuf, stats, outP, Xh, Bp);
  softmax_mul<<<divup_i((long)PREDN * 32, 256), 256, 0, stream>>>(h_uS, goutB, socbuf, PREDN);
  run_mlp(stream, socbuf, h_uS, PREDN, MW(4), MB(4), MG(4), MBe(4), fsbuf, stats, outS, Xh, Bp);

  (void)n_in; (void)out_size; (void)ws_size;
}
